// FlowFixEquivariantModel_70317204570121
// MI455X (gfx1250) — compile-verified, matrix operand order checked by reading, not run on hardware
//
#include <hip/hip_runtime.h>

typedef __attribute__((ext_vector_type(16))) _Float16 v16h;
typedef __attribute__((ext_vector_type(8)))  float    v8f;

#define NLIG 2048
#define NPRO 4096
#define NNODE 6144
#define NEDGE 8192
#define NLAYER 6
#define NWC 11520
#define I48f 0.14433756729740643f
#define I16f 0.25f
#define EPSf 1e-5f

__device__ __forceinline__ float silu(float x){ return x / (1.f + __expf(-x)); }

// ---------------- utility ----------------
__global__ void zero_kernel(float* __restrict__ p, int n){
  int i = blockIdx.x*blockDim.x + threadIdx.x;
  int st = gridDim.x*blockDim.x;
  for (; i<n; i+=st) p[i]=0.f;
}

__global__ void deg_kernel(const int* __restrict__ edst, float* __restrict__ deg){
  int e = blockIdx.x*blockDim.x + threadIdx.x;
  if (e<NEDGE) atomicAdd(&deg[edst[e]+NLIG], 1.f);
}

// ---------------- time embedding + edge-feature base ----------------
__global__ __launch_bounds__(128) void time_kernel(
  const float* __restrict__ t,
  const float* __restrict__ te_w1, const float* __restrict__ te_b1,
  const float* __restrict__ te_w2, const float* __restrict__ te_b2,
  const float* __restrict__ lng, const float* __restrict__ lnb,
  const float* __restrict__ ee_w1, const float* __restrict__ ee_b1,
  float* __restrict__ temb, float* __restrict__ efbase)
{
  __shared__ float emb[64], h[128], o[64], tb[64];
  int tid = threadIdx.x;
  float tv = t[0];
  if (tid < 64){
    int j = (tid<32)? tid : tid-32;
    float fr = expf((float)j * (-9.210340371976184f/31.f)); // -ln(1e4)/31
    emb[tid] = (tid<32)? sinf(tv*fr) : cosf(tv*fr);
  }
  __syncthreads();
  float a = te_b1[tid];
  for (int j=0;j<64;j++) a += emb[j]*te_w1[j*128+tid];
  h[tid] = silu(a);
  __syncthreads();
  if (tid < 64){
    float c = te_b2[tid];
    for (int k=0;k<128;k++) c += h[k]*te_w2[k*64+tid];
    o[tid]=c;
  }
  __syncthreads();
  if (tid < 64){
    float mu=0; for (int j=0;j<64;j++) mu+=o[j]; mu *= (1.f/64.f);
    float var=0; for (int j=0;j<64;j++){ float d=o[j]-mu; var+=d*d; } var *= (1.f/64.f);
    float val = lng[tid]*(o[tid]-mu)*rsqrtf(var+EPSf)+lnb[tid];
    tb[tid]=val; temb[tid]=val;
  }
  __syncthreads();
  if (tid < 32){
    float c = ee_b1[tid];
    for (int j=0;j<64;j++) c += tb[j]*ee_w1[(1+j)*32+tid];
    efbase[tid]=c;
  }
}

// ---------------- protein / ligand encoders ----------------
__global__ __launch_bounds__(128) void encode_kernel(
  const float* __restrict__ X, int fin,
  const float* __restrict__ w1, const float* __restrict__ b1,
  const float* __restrict__ lng, const float* __restrict__ lnb,
  const float* __restrict__ w2, const float* __restrict__ b2,
  float* __restrict__ feat, int base)
{
  __shared__ float xl[96], h[128], red[128];
  int n = blockIdx.x, tid = threadIdx.x;
  for (int i=tid;i<fin;i+=128) xl[i]=X[(size_t)n*fin+i];
  __syncthreads();
  float a = b1[tid];
  for (int i=0;i<fin;i++) a += xl[i]*w1[i*128+tid];
  red[tid]=a; __syncthreads();
  for (int st=64; st>0; st>>=1){ if (tid<st) red[tid]+=red[tid+st]; __syncthreads(); }
  float mu = red[0]*(1.f/128.f);
  __syncthreads();
  float dd = a-mu; red[tid]=dd*dd; __syncthreads();
  for (int st=64; st>0; st>>=1){ if (tid<st) red[tid]+=red[tid+st]; __syncthreads(); }
  float var = red[0]*(1.f/128.f);
  float lnv = lng[tid]*dd*rsqrtf(var+EPSf)+lnb[tid];
  __syncthreads();
  h[tid]=silu(lnv);
  __syncthreads();
  if (tid < 48){
    float c = b2[tid];
    for (int i=0;i<128;i++) c += h[i]*w2[i*48+tid];
    feat[(size_t)(base+n)*48+tid]=c;
  }
}

// ---------------- node init: s = ((feat@sp_w+sp_b)@emb_w)*I48 ----------------
__global__ __launch_bounds__(64) void node_init_kernel(
  const float* __restrict__ feat, const float* __restrict__ sp_w,
  const float* __restrict__ sp_b, const float* __restrict__ emb_w,
  float* __restrict__ s)
{
  __shared__ float fl[48], comb[48];
  int n=blockIdx.x, tid=threadIdx.x;
  if (tid<48) fl[tid]=feat[(size_t)n*48+tid];
  __syncthreads();
  if (tid<48){
    float a=sp_b[tid];
    for (int i=0;i<48;i++) a+=fl[i]*sp_w[i*48+tid];
    comb[tid]=a;
  }
  __syncthreads();
  if (tid<48){
    float b=0;
    for (int i=0;i<48;i++) b+=comb[i]*emb_w[i*48+tid];
    s[(size_t)n*48+tid]=b*I48f;
  }
}

// ---------------- edge geometry + edge features ----------------
__global__ void edge_kernel(
  const float* __restrict__ lc, const float* __restrict__ pc,
  const int* __restrict__ esrc, const int* __restrict__ edst,
  const float* __restrict__ ee_w1, const float* __restrict__ ee_b2,
  const float* __restrict__ ee_w2, const float* __restrict__ efbase,
  float* __restrict__ ne, float* __restrict__ ef)
{
  int e = blockIdx.x*blockDim.x+threadIdx.x;
  if (e>=NEDGE) return;
  int sidx = esrc[e], didx = edst[e];
  float vx = lc[sidx*3+0]-pc[didx*6+0];
  float vy = lc[sidx*3+1]-pc[didx*6+1];
  float vz = lc[sidx*3+2]-pc[didx*6+2];
  float dist = sqrtf(vx*vx+vy*vy+vz*vz+1e-12f);
  float inv = 1.f/dist;
  ne[e*3+0]=vx*inv; ne[e*3+1]=vy*inv; ne[e*3+2]=vz*inv;
  float hb[32];
  #pragma unroll
  for (int c=0;c<32;c++) hb[c]=silu(dist*ee_w1[c]+efbase[c]);
  for (int o=0;o<32;o++){
    float a=ee_b2[o];
    #pragma unroll
    for (int c=0;c<32;c++) a+=hb[c]*ee_w2[c*32+o];
    ef[(size_t)e*32+o]=a;
  }
}

// ---------------- per-layer edge MLP -> f16 activations ----------------
__global__ __launch_bounds__(128) void edge_mlp_kernel(
  const float* __restrict__ ef,
  const float* __restrict__ w1, const float* __restrict__ b1,
  const float* __restrict__ w2, const float* __restrict__ b2,
  _Float16* __restrict__ m2h)
{
  __shared__ float efl[32], h1[128];
  int e=blockIdx.x, k=threadIdx.x;
  if (k<32) efl[k]=ef[(size_t)e*32+k];
  __syncthreads();
  float a=b1[k];
  #pragma unroll
  for (int i=0;i<32;i++) a += efl[i]*w1[i*128+k];
  h1[k]=silu(a);
  __syncthreads();
  float c=b2[k];
  for (int i=0;i<128;i++) c += h1[i]*w2[i*128+k];
  m2h[(size_t)e*128+k] = (_Float16)silu(c);
}

// ---------------- conv_w3 -> f16, pre-swizzled into WMMA B fragments ----------------
// layout: [cb(720)][kc(4)][lane(32)] x v16h(16 halfs)
__global__ void w3_swizzle_kernel(const float* __restrict__ w3, _Float16* __restrict__ w3t)
{
  int idx = blockIdx.x*blockDim.x+threadIdx.x;
  if (idx >= 720*4*32*16) return;
  int h    = idx & 15;
  int lane = (idx>>4)&31;
  int kc   = (idx>>9)&3;
  int cb   = idx>>11;
  int g = lane>>4, nn = lane&15;
  int K = kc*32 + (h<8 ? 8*g+h : 8*g+8+h);
  w3t[idx] = (_Float16)w3[(size_t)K*NWC + cb*16 + nn];
}

// ---------------- fused generate(WMMA) + apply + scatter ----------------
__global__ __launch_bounds__(32) void gen_apply_kernel(
    const _Float16* __restrict__ m2h,
    const v16h*  __restrict__ w3tv,
    const float* __restrict__ b3,
    const float* __restrict__ s,
    const float* __restrict__ v,
    const float* __restrict__ w,
    const float* __restrict__ ne,
    const int*   __restrict__ esrc,
    const int*   __restrict__ edst,
    float* __restrict__ Sb, float* __restrict__ Vb, float* __restrict__ Wb)
{
  __shared__ float sSE[16*48];
  __shared__ float sWN[16*16];
  // merged factor arrays indexed by path id (avoids LDS pointer tables):
  // sV3[pp]: pp=0 -> ve, 1 -> cv, 2 -> qv   ; each [16 edges][48][3]
  // sW3[pp]: pp=0 -> we, 1 -> cw, 2 -> qw   ; each [16 edges][16][3]
  __shared__ float sV3[3*16*48*3];
  __shared__ float sW3[3*16*16*3];
  __shared__ float sN[16*3];
  __shared__ int   sSRC[16], sDST[16];

  const int lane = threadIdx.x;
  const int e0 = blockIdx.x*16;
  const int g  = lane>>4;
  const int nn = lane & 15;

  if (lane < 16){
    int e = e0 + lane;
    sSRC[lane] = esrc[e];
    sDST[lane] = edst[e] + NLIG;
    sN[lane*3+0]=ne[e*3+0];
    sN[lane*3+1]=ne[e*3+1];
    sN[lane*3+2]=ne[e*3+2];
  }
  __syncthreads();

  for (int idx=lane; idx<16*48; idx+=32){
    int el=idx/48, i=idx-el*48;
    int node = sSRC[el];
    sSE[idx] = s[(size_t)node*48+i];
    const float* vp = v + ((size_t)node*48+i)*3;
    float vx=vp[0], vy=vp[1], vz=vp[2];
    float nx=sN[el*3+0], ny=sN[el*3+1], nz=sN[el*3+2];
    float dt=vx*nx+vy*ny+vz*nz;
    int b = idx*3;
    sV3[0*2304+b+0]=vx;              sV3[0*2304+b+1]=vy;              sV3[0*2304+b+2]=vz;
    sV3[1*2304+b+0]=vy*nz-vz*ny;     sV3[1*2304+b+1]=vz*nx-vx*nz;     sV3[1*2304+b+2]=vx*ny-vy*nx;
    sV3[2*2304+b+0]=nx*dt-vx*(1.f/3.f);
    sV3[2*2304+b+1]=ny*dt-vy*(1.f/3.f);
    sV3[2*2304+b+2]=nz*dt-vz*(1.f/3.f);
  }
  for (int idx=lane; idx<16*16; idx+=32){
    int el=idx>>4, i=idx&15;
    int node = sSRC[el];
    const float* wp = w + ((size_t)node*16+i)*3;
    float wx=wp[0], wy=wp[1], wz=wp[2];
    float nx=sN[el*3+0], ny=sN[el*3+1], nz=sN[el*3+2];
    float dt=wx*nx+wy*ny+wz*nz;
    int b = idx*3;
    sW3[0*768+b+0]=wx;               sW3[0*768+b+1]=wy;               sW3[0*768+b+2]=wz;
    sW3[1*768+b+0]=wy*nz-wz*ny;      sW3[1*768+b+1]=wz*nx-wx*nz;      sW3[1*768+b+2]=wx*ny-wy*nx;
    sW3[2*768+b+0]=nx*dt-wx*(1.f/3.f);
    sW3[2*768+b+1]=ny*dt-wy*(1.f/3.f);
    sW3[2*768+b+2]=nz*dt-wz*(1.f/3.f);
    sWN[idx]=dt;
  }
  __syncthreads();

  // A fragments: 16 edges x 128 K, f16 (rows = lane%16)
  v16h A[4];
  {
    const _Float16* mp = m2h + (size_t)(e0 + nn)*128;
    #pragma unroll
    for (int kc=0;kc<4;kc++){
      #pragma unroll
      for (int h=0;h<16;h++){
        int K = kc*32 + (h<8 ? 8*g + h : 8*g + 8 + h);
        A[kc][h] = mp[K];
      }
    }
  }

  auto gen = [&](int cb) -> v8f {
    const v16h* bp = w3tv + (size_t)cb*128 + lane;
    __builtin_prefetch((const void*)(bp + 128), 0, 1);
    float bias = b3[(cb<<4) + nn];
    v8f d;
    #pragma unroll
    for (int r=0;r<8;r++) d[r]=bias;
    #pragma unroll
    for (int kc=0;kc<4;kc++)
      d = __builtin_amdgcn_wmma_f32_16x16x32_f16(false, A[kc], false, bp[kc*32],
                                                 (short)0, d, false, false);
    return d;
  };

  float accS[3][8], accV[3][3][8], accW[3][8];
  #pragma unroll
  for (int a=0;a<3;a++)
    #pragma unroll
    for (int r=0;r<8;r++){
      accS[a][r]=0.f; accW[a][r]=0.f;
      accV[a][0][r]=0.f; accV[a][1][r]=0.f; accV[a][2][r]=0.f;
    }

  // p0: Wss (48,48) -> m_s, f = se*I48, cb base 0
  for (int i=0;i<48;i++){
    float f[8];
    #pragma unroll
    for (int r=0;r<8;r++) f[r]=sSE[(8*g+r)*48+i]*I48f;
    for (int ob=0;ob<3;ob++){
      v8f d = gen(i*3+ob);
      #pragma unroll
      for (int r=0;r<8;r++) accS[ob][r]+=d[r]*f[r];
    }
  }
  // p6: Wws (16,48) -> m_s, f = wn*I16, cb base 640
  for (int i=0;i<16;i++){
    float f[8];
    #pragma unroll
    for (int r=0;r<8;r++) f[r]=sWN[(8*g+r)*16+i]*I16f;
    for (int ob=0;ob<3;ob++){
      v8f d = gen(640+i*3+ob);
      #pragma unroll
      for (int r=0;r<8;r++) accS[ob][r]+=d[r]*f[r];
    }
  }
  // p2/p3/p4: Wv0/Wv1/Wv2 (48,48) -> m_v, f = {ve,cv,qv}*I48, cb 192/336/480
  for (int pp=0;pp<3;pp++){
    const int cb0  = 192 + 144*pp;
    const int base = pp*2304;
    for (int i=0;i<48;i++){
      float f0[8],f1[8],f2[8];
      #pragma unroll
      for (int r=0;r<8;r++){
        int b = base + ((8*g+r)*48+i)*3;
        f0[r]=sV3[b]*I48f; f1[r]=sV3[b+1]*I48f; f2[r]=sV3[b+2]*I48f;
      }
      for (int ob=0;ob<3;ob++){
        v8f d = gen(cb0+i*3+ob);
        #pragma unroll
        for (int r=0;r<8;r++){
          accV[ob][0][r]+=d[r]*f0[r];
          accV[ob][1][r]+=d[r]*f1[r];
          accV[ob][2][r]+=d[r]*f2[r];
        }
      }
    }
  }
  // p1: Wsw (48,16) -> m_w, f_d = se*n_d*I48, cb 144+i
  for (int i=0;i<48;i++){
    float f0[8],f1[8],f2[8];
    #pragma unroll
    for (int r=0;r<8;r++){
      int e=8*g+r; float sv=sSE[e*48+i]*I48f;
      f0[r]=sv*sN[e*3+0]; f1[r]=sv*sN[e*3+1]; f2[r]=sv*sN[e*3+2];
    }
    v8f d = gen(144+i);
    #pragma unroll
    for (int r=0;r<8;r++){
      accW[0][r]+=d[r]*f0[r]; accW[1][r]+=d[r]*f1[r]; accW[2][r]+=d[r]*f2[r];
    }
  }
  // p5/p7/p8: Ww0/Ww1/Ww2 (16,16) -> m_w, f = {we,cw,qw}*I16, cb 624/688/704
  {
    const int wcb[3]={624,688,704};
    for (int pp=0;pp<3;pp++){
      const int cb0  = wcb[pp];
      const int base = pp*768;
      for (int i=0;i<16;i++){
        float f0[8],f1[8],f2[8];
        #pragma unroll
        for (int r=0;r<8;r++){
          int b = base + ((8*g+r)*16+i)*3;
          f0[r]=sW3[b]*I16f; f1[r]=sW3[b+1]*I16f; f2[r]=sW3[b+2]*I16f;
        }
        v8f d = gen(cb0+i);
        #pragma unroll
        for (int r=0;r<8;r++){
          accW[0][r]+=d[r]*f0[r]; accW[1][r]+=d[r]*f1[r]; accW[2][r]+=d[r]*f2[r];
        }
      }
    }
  }

  // scatter into node message buffers (segment-sum via f32 atomics)
  #pragma unroll
  for (int r=0;r<8;r++){
    int node = sDST[8*g+r];
    #pragma unroll
    for (int ob=0;ob<3;ob++)
      atomicAdd(&Sb[(size_t)node*48 + ob*16 + nn], accS[ob][r]);
    #pragma unroll
    for (int ob=0;ob<3;ob++)
      #pragma unroll
      for (int dd=0;dd<3;dd++)
        atomicAdd(&Vb[((size_t)node*48 + ob*16 + nn)*3 + dd], accV[ob][dd][r]);
    #pragma unroll
    for (int dd=0;dd<3;dd++)
      atomicAdd(&Wb[((size_t)node*16 + nn)*3 + dd], accW[dd][r]);
  }
}

// ---------------- node mix: a_* = msg/deg + self-interaction ----------------
__global__ __launch_bounds__(64) void node_mix_kernel(
  const float* __restrict__ s, const float* __restrict__ v, const float* __restrict__ w,
  const float* __restrict__ deg,
  const float* __restrict__ Sb, const float* __restrict__ Vb, const float* __restrict__ Wb,
  const float* __restrict__ si_ws, const float* __restrict__ si_wv, const float* __restrict__ si_ww,
  float* __restrict__ a_s, float* __restrict__ a_v, float* __restrict__ a_w)
{
  __shared__ float sl[48], vl[144], wl[48];
  int n=blockIdx.x, t=threadIdx.x;
  if (t<48) sl[t]=s[(size_t)n*48+t];
  for (int i=t;i<144;i+=64) vl[i]=v[(size_t)n*144+i];
  if (t<48) wl[t]=w[(size_t)n*48+t];
  __syncthreads();
  float inv = 1.f/fmaxf(deg[n],1.f);
  if (t<48){
    int o=t;
    float t0=0;
    for (int i=0;i<48;i++) t0 += sl[i]*si_ws[i*48+o];
    a_s[(size_t)n*48+o] = Sb[(size_t)n*48+o]*inv + t0*I48f;

    float u0=0,u1=0,u2=0;
    for (int i=0;i<48;i++){
      float wv=si_wv[i*48+o];
      u0+=wv*vl[i*3]; u1+=wv*vl[i*3+1]; u2+=wv*vl[i*3+2];
    }
    size_t bi = ((size_t)n*48+o)*3;
    a_v[bi+0]=Vb[bi+0]*inv + u0*I48f;
    a_v[bi+1]=Vb[bi+1]*inv + u1*I48f;
    a_v[bi+2]=Vb[bi+2]*inv + u2*I48f;

    if (o<16){
      float q0=0,q1=0,q2=0;
      for (int i=0;i<16;i++){
        float ww=si_ww[i*16+o];
        q0+=ww*wl[i*3]; q1+=ww*wl[i*3+1]; q2+=ww*wl[i*3+2];
      }
      size_t wi = ((size_t)n*16+o)*3;
      a_w[wi+0]=Wb[wi+0]*inv + q0*I16f;
      a_w[wi+1]=Wb[wi+1]*inv + q1*I16f;
      a_w[wi+2]=Wb[wi+2]*inv + q2*I16f;
    }
  }
}

// ---------------- batch-norm statistics (one block per channel task) ----------------
__global__ __launch_bounds__(256) void stats_kernel(
  const float* __restrict__ a_s, const float* __restrict__ a_v, const float* __restrict__ a_w,
  float* __restrict__ mean_s, float* __restrict__ var_s,
  float* __restrict__ msq_v, float* __restrict__ msq_w)
{
  __shared__ float r1[256], r2[256];
  int b=blockIdx.x, t=threadIdx.x;
  float s1=0, s2=0;
  if (b<48){
    for (int n=t;n<NNODE;n+=256){ float x=a_s[(size_t)n*48+b]; s1+=x; s2+=x*x; }
  } else if (b<96){
    int o=b-48;
    for (int n=t;n<NNODE;n+=256){
      const float* pv=&a_v[((size_t)n*48+o)*3];
      s1 += pv[0]*pv[0]+pv[1]*pv[1]+pv[2]*pv[2];
    }
  } else {
    int o=b-96;
    for (int n=t;n<NNODE;n+=256){
      const float* pw=&a_w[((size_t)n*16+o)*3];
      s1 += pw[0]*pw[0]+pw[1]*pw[1]+pw[2]*pw[2];
    }
  }
  r1[t]=s1; r2[t]=s2; __syncthreads();
  for (int st=128;st>0;st>>=1){ if (t<st){ r1[t]+=r1[t+st]; r2[t]+=r2[t+st]; } __syncthreads(); }
  if (t==0){
    const float invN = 1.f/(float)NNODE;
    if (b<48){ float mu=r1[0]*invN; mean_s[b]=mu; var_s[b]=r2[0]*invN - mu*mu; }
    else if (b<96) msq_v[b-48]=r1[0]*invN;
    else           msq_w[b-96]=r1[0]*invN;
  }
}

// ---------------- apply norms + residual update ----------------
__global__ void norm_kernel(
  const float* __restrict__ a_s, const float* __restrict__ a_v, const float* __restrict__ a_w,
  const float* __restrict__ mean_s, const float* __restrict__ var_s,
  const float* __restrict__ msq_v, const float* __restrict__ msq_w,
  const float* __restrict__ gs, const float* __restrict__ bs,
  const float* __restrict__ gv, const float* __restrict__ gw,
  float* __restrict__ s, float* __restrict__ v, float* __restrict__ w)
{
  int idx=blockIdx.x*blockDim.x+threadIdx.x;
  if (idx>=NNODE*48) return;
  int n=idx/48, o=idx-n*48;
  s[idx] += gs[o]*(a_s[idx]-mean_s[o])*rsqrtf(var_s[o]+EPSf)+bs[o];
  float fv = gv[o]*rsqrtf(msq_v[o]+EPSf);
  size_t bi=(size_t)idx*3;
  v[bi+0]+=a_v[bi+0]*fv; v[bi+1]+=a_v[bi+1]*fv; v[bi+2]+=a_v[bi+2]*fv;
  if (o<16){
    size_t wi=((size_t)n*16+o)*3;
    float fw = gw[o]*rsqrtf(msq_w[o]+EPSf);
    w[wi+0]+=a_w[wi+0]*fw; w[wi+1]+=a_w[wi+1]*fw; w[wi+2]+=a_w[wi+2]*fw;
  }
}

// ---------------- output head ----------------
__global__ __launch_bounds__(256) void output_kernel(
  const float* __restrict__ s, const float* __restrict__ v,
  const float* __restrict__ out_ws, const float* __restrict__ out_wv,
  float* __restrict__ out)
{
  __shared__ float sl[48], vl[144];
  int n=blockIdx.x, t=threadIdx.x;
  if (t<48) sl[t]=s[(size_t)n*48+t];
  if (t>=64 && t<64+144) vl[t-64]=v[(size_t)n*144+(t-64)];
  __syncthreads();
  float a=0;
  for (int i=0;i<48;i++) a+=sl[i]*out_ws[i*256+t];
  out[(size_t)n*259 + 3 + t] = a*I48f;
  if (t<3){
    float b=0;
    for (int i=0;i<48;i++){
      float gsum = out_wv[i*3+0]+out_wv[i*3+1]+out_wv[i*3+2];
      b += gsum*vl[i*3+t];
    }
    out[(size_t)n*259 + t] = b*(I48f/3.f);
  }
}

// ================= host orchestration =================
extern "C" void kernel_launch(void* const* d_in, const int* in_sizes, int n_in,
                              void* d_out, int out_size, void* d_ws, size_t ws_size,
                              hipStream_t stream)
{
  (void)in_sizes; (void)n_in; (void)out_size; (void)ws_size;
  const float* protein_coords   = (const float*)d_in[0];
  const float* protein_features = (const float*)d_in[1];
  const float* ligand_coords    = (const float*)d_in[2];
  const float* ligand_features  = (const float*)d_in[3];
  const float* t      = (const float*)d_in[4];
  const float* te_w1  = (const float*)d_in[5];
  const float* te_b1  = (const float*)d_in[6];
  const float* te_w2  = (const float*)d_in[7];
  const float* te_b2  = (const float*)d_in[8];
  const float* te_ln_g= (const float*)d_in[9];
  const float* te_ln_b= (const float*)d_in[10];
  const float* pe_w1  = (const float*)d_in[11];
  const float* pe_b1  = (const float*)d_in[12];
  const float* pe_ln_g= (const float*)d_in[13];
  const float* pe_ln_b= (const float*)d_in[14];
  const float* pe_w2  = (const float*)d_in[15];
  const float* pe_b2  = (const float*)d_in[16];
  const float* le_w1  = (const float*)d_in[17];
  const float* le_b1  = (const float*)d_in[18];
  const float* le_ln_g= (const float*)d_in[19];
  const float* le_ln_b= (const float*)d_in[20];
  const float* le_w2  = (const float*)d_in[21];
  const float* le_b2  = (const float*)d_in[22];
  const float* sp_w   = (const float*)d_in[23];
  const float* sp_b   = (const float*)d_in[24];
  const float* emb_w  = (const float*)d_in[25];
  const float* ee_w1  = (const float*)d_in[26];
  const float* ee_b1  = (const float*)d_in[27];
  const float* ee_w2  = (const float*)d_in[28];
  const float* ee_b2  = (const float*)d_in[29];
  const float* conv_w1= (const float*)d_in[30];
  const float* conv_b1= (const float*)d_in[31];
  const float* conv_w2= (const float*)d_in[32];
  const float* conv_b2= (const float*)d_in[33];
  const float* conv_w3= (const float*)d_in[34];
  const float* conv_b3= (const float*)d_in[35];
  const float* bn_gs  = (const float*)d_in[36];
  const float* bn_bs  = (const float*)d_in[37];
  const float* bn_gv  = (const float*)d_in[38];
  const float* bn_gw  = (const float*)d_in[39];
  const float* si_ws  = (const float*)d_in[40];
  const float* si_wv  = (const float*)d_in[41];
  const float* si_ww  = (const float*)d_in[42];
  const float* out_ws = (const float*)d_in[43];
  const float* out_wv = (const float*)d_in[44];
  const int* edge_src = (const int*)d_in[45];
  const int* edge_dst = (const int*)d_in[46];
  float* out = (float*)d_out;

  float* p = (float*)d_ws;
  auto take = [&](size_t n){ float* q=p; p+=n; return q; };
  float* temb   = take(64);
  float* efbase = take(64);
  float* sS   = take((size_t)NNODE*48);
  float* vV   = take((size_t)NNODE*144);
  float* wW   = take((size_t)NNODE*48);
  float* deg  = take(NNODE);
  float* ne   = take((size_t)NEDGE*3);
  float* ef   = take((size_t)NEDGE*32);
  float* feat = take((size_t)NNODE*48);
  float* Sb   = take((size_t)NNODE*48);
  float* Vb   = take((size_t)NNODE*144);
  float* Wb   = take((size_t)NNODE*48);
  float* a_s  = take((size_t)NNODE*48);
  float* a_v  = take((size_t)NNODE*144);
  float* a_w  = take((size_t)NNODE*48);
  float* mean_s = take(64);
  float* var_s  = take(64);
  float* msq_v  = take(64);
  float* msq_w  = take(64);
  _Float16* m2h = (_Float16*)take((size_t)NEDGE*64);       // E*128 halfs
  _Float16* w3t = (_Float16*)take((size_t)720*4*32*16/2);  // swizzled f16 weights

  time_kernel<<<1,128,0,stream>>>(t, te_w1, te_b1, te_w2, te_b2, te_ln_g, te_ln_b,
                                  ee_w1, ee_b1, temb, efbase);
  encode_kernel<<<NPRO,128,0,stream>>>(protein_features, 96, pe_w1, pe_b1, pe_ln_g, pe_ln_b,
                                       pe_w2, pe_b2, feat, NLIG);
  encode_kernel<<<NLIG,128,0,stream>>>(ligand_features, 52, le_w1, le_b1, le_ln_g, le_ln_b,
                                       le_w2, le_b2, feat, 0);
  node_init_kernel<<<NNODE,64,0,stream>>>(feat, sp_w, sp_b, emb_w, sS);
  zero_kernel<<<1024,256,0,stream>>>(vV, NNODE*192);   // v and w (contiguous)
  zero_kernel<<<64,256,0,stream>>>(deg, NNODE);
  deg_kernel<<<(NEDGE+255)/256,256,0,stream>>>(edge_dst, deg);
  edge_kernel<<<(NEDGE+127)/128,128,0,stream>>>(ligand_coords, protein_coords, edge_src, edge_dst,
                                                ee_w1, ee_b2, ee_w2, efbase, ne, ef);

  for (int L=0; L<NLAYER; L++){
    zero_kernel<<<1024,256,0,stream>>>(Sb, NNODE*240);   // Sb,Vb,Wb contiguous
    w3_swizzle_kernel<<<(720*4*32*16+255)/256,256,0,stream>>>(conv_w3 + (size_t)L*128*NWC, w3t);
    edge_mlp_kernel<<<NEDGE,128,0,stream>>>(ef, conv_w1+(size_t)L*32*128, conv_b1+L*128,
                                            conv_w2+(size_t)L*128*128, conv_b2+L*128, m2h);
    gen_apply_kernel<<<NEDGE/16,32,0,stream>>>(m2h, (const v16h*)w3t, conv_b3+(size_t)L*NWC,
                                               sS, vV, wW, ne, edge_src, edge_dst, Sb, Vb, Wb);
    node_mix_kernel<<<NNODE,64,0,stream>>>(sS, vV, wW, deg, Sb, Vb, Wb,
                                           si_ws+(size_t)L*2304, si_wv+(size_t)L*2304,
                                           si_ww+(size_t)L*256, a_s, a_v, a_w);
    stats_kernel<<<112,256,0,stream>>>(a_s, a_v, a_w, mean_s, var_s, msq_v, msq_w);
    norm_kernel<<<(NNODE*48+255)/256,256,0,stream>>>(a_s, a_v, a_w, mean_s, var_s, msq_v, msq_w,
                                                     bn_gs+L*48, bn_bs+L*48, bn_gv+L*48, bn_gw+L*16,
                                                     sS, vV, wW);
  }
  output_kernel<<<NLIG,256,0,stream>>>(sS, vV, out_ws, out_wv, out);
}
